// PatchQuantumGenerator_67070209294473
// MI455X (gfx1250) — matrix operand order repaired and checked by reading, not compiled, not run on hardware
//
#include <hip/hip_runtime.h>
#include <hip/hip_fp16.h>
#include <hip/hip_bf16.h>

typedef __attribute__((ext_vector_type(16))) _Float16 v16h;
typedef __attribute__((ext_vector_type(8)))  _Float16 v8h;
typedef __attribute__((ext_vector_type(8)))  float    v8f;

#define BATCH 2048
#define NQ    10
#define NGEN  4
#define DEPTH 6
#define KPAD  1056   // 1024 qf + 10 labels + 22 zero pad
#define PI_F  3.14159265358979323846f

// ---------------------------------------------------------------------------
// Quantum circuit: one wave32 simulates one (batch, generator) circuit.
// Statevector: 1024 complex amps -> 32 per lane (re[32], im[32] in VGPRs).
// Index bits [4:0] = lane, bits [9:5] = register. Qubit q <-> bit (9-q).
// ---------------------------------------------------------------------------
template <int BIT>
__device__ __forceinline__ void ry_gate(float (&re)[32], float (&im)[32],
                                        float th, int lane) {
  float s, c;
  __sincosf(0.5f * th, &s, &c);
  if constexpr (BIT >= 5) {
    constexpr int m = 1 << (BIT - 5);
#pragma unroll
    for (int r = 0; r < 32; ++r) {
      if (!(r & m)) {
        float t = re[r];
        re[r]     = c * t - s * re[r | m];
        re[r | m] = s * t + c * re[r | m];
        float u = im[r];
        im[r]     = c * u - s * im[r | m];
        im[r | m] = s * u + c * im[r | m];
      }
    }
  } else {
    // pair across lanes: a' = c*a + sgn*s*partner   (sgn = -1 if bit==0)
    const float ss = (((lane >> BIT) & 1) ? s : -s);
#pragma unroll
    for (int r = 0; r < 32; ++r) {
      float pr = __shfl_xor(re[r], 1 << BIT, 32);
      float pi = __shfl_xor(im[r], 1 << BIT, 32);
      re[r] = c * re[r] + ss * pr;
      im[r] = c * im[r] + ss * pi;
    }
  }
}

template <int BIT>
__device__ __forceinline__ void rz_gate(float (&re)[32], float (&im)[32],
                                        float th, int lane) {
  float s, c;
  __sincosf(0.5f * th, &s, &c);
  if constexpr (BIT >= 5) {
    constexpr int m = 1 << (BIT - 5);
#pragma unroll
    for (int r = 0; r < 32; ++r) {
      float sg = (r & m) ? s : -s;
      float tr = re[r] * c - sg * im[r];
      im[r]    = im[r] * c + sg * re[r];
      re[r]    = tr;
    }
  } else {
    const float sg = (((lane >> BIT) & 1) ? s : -s);
#pragma unroll
    for (int r = 0; r < 32; ++r) {
      float tr = re[r] * c - sg * im[r];
      im[r]    = im[r] * c + sg * re[r];
      re[r]    = tr;
    }
  }
}

#define FOR10(M) M(0) M(1) M(2) M(3) M(4) M(5) M(6) M(7) M(8) M(9)
#define NOISE_RY(q) ry_gate<9 - (q)>(re, im, nz[(q)], lane);
#define ENC_RYRZ(q)                                  \
  ry_gate<9 - (q)>(re, im, an[(q)], lane);           \
  rz_gate<9 - (q)>(re, im, an[10 + (q)], lane);
#define LAYER_RY(q) ry_gate<9 - (q)>(re, im, w[(q)], lane);

__global__ __launch_bounds__(256) void circuit_kernel(
    const float* __restrict__ noise,    // (2048,10)
    const float* __restrict__ angles,   // (2048,20)
    const float* __restrict__ qparams,  // (4,60)
    _Float16* __restrict__ xpad)        // (2048,KPAD): writes cols g*256..+255
{
  const int wave = (blockIdx.x * blockDim.x + threadIdx.x) >> 5;  // 0..8191
  const int lane = threadIdx.x & 31;
  const int b = wave >> 2;
  const int g = wave & 3;

  const float* nz = noise + b * NQ;
  const float* an = angles + b * 2 * NQ;
  const float* qp = qparams + g * DEPTH * NQ;

  float re[32], im[32];
#pragma unroll
  for (int r = 0; r < 32; ++r) { re[r] = 0.f; im[r] = 0.f; }
  re[0] = (lane == 0) ? 1.f : 0.f;   // |0...0>

  FOR10(NOISE_RY)                    // RY(noise_q) on every wire
  FOR10(ENC_RYRZ)                    // RY(ang_q), RZ(ang_{10+q})

  // Per-lane sign of the fused CZ chain restricted to lane bits:
  // pairs (4,3)(3,2)(2,1)(1,0); mixed pair (reg bit0, lane bit4) handled below.
  {
    const int l = lane;
    const int cl = ((l >> 4) & (l >> 3) & 1) + ((l >> 3) & (l >> 2) & 1) +
                   ((l >> 2) & (l >> 1) & 1) + ((l >> 1) & l & 1);
    const float sgnL  = (cl & 1) ? -1.f : 1.f;
    const float feven = sgnL;
    const float fodd  = (lane & 16) ? -sgnL : sgnL;

    for (int ld = 0; ld < DEPTH; ++ld) {
      const float* w = qp + ld * NQ;
      FOR10(LAYER_RY)
      // Fused CZ chain (bits (9,8)(8,7)...(1,0)) as one diagonal sign pass.
#pragma unroll
      for (int r = 0; r < 32; ++r) {
        const int cr = ((r >> 4) & (r >> 3) & 1) + ((r >> 3) & (r >> 2) & 1) +
                       ((r >> 2) & (r >> 1) & 1) + ((r >> 1) & r & 1);
        float f = (cr & 1) ? -1.f : 1.f;
        f *= (r & 1) ? fodd : feven;
        re[r] *= f;
        im[r] *= f;
      }
    }
  }

  // probs for indices 0..255 (regs 0..7); p/sum then /max == p/max (sum cancels)
  float p[8], m = 0.f;
#pragma unroll
  for (int r = 0; r < 8; ++r) {
    p[r] = re[r] * re[r] + im[r] * im[r];
    m = fmaxf(m, p[r]);
  }
#pragma unroll
  for (int off = 16; off >= 1; off >>= 1) m = fmaxf(m, __shfl_xor(m, off, 32));
  const float rinv = 1.0f / m;

  _Float16* dst = xpad + (size_t)b * KPAD + g * 256 + lane;
#pragma unroll
  for (int r = 0; r < 8; ++r) dst[r * 32] = (_Float16)(p[r] * rinv);
}

// ---------------------------------------------------------------------------
// Encoder MLP: labels(10) -> 32 -> 64 -> 20 angles; also writes label + zero
// pad columns of Xpad. One thread per batch row; weights staged in LDS.
// ---------------------------------------------------------------------------
__global__ __launch_bounds__(256) void encoder_kernel(
    const float* __restrict__ labels,
    const float* __restrict__ ew1, const float* __restrict__ eb1,
    const float* __restrict__ ew2, const float* __restrict__ eb2,
    const float* __restrict__ ew3, const float* __restrict__ eb3,
    float* __restrict__ angles, _Float16* __restrict__ xpad)
{
  __shared__ float sw1[32 * 10], sb1[32];
  __shared__ float sw2[64 * 32], sb2[64];
  __shared__ float sw3[20 * 64], sb3[20];
  const int tid = threadIdx.x;
  for (int i = tid; i < 320;  i += 256) sw1[i] = ew1[i];
  for (int i = tid; i < 32;   i += 256) sb1[i] = eb1[i];
  for (int i = tid; i < 2048; i += 256) sw2[i] = ew2[i];
  for (int i = tid; i < 64;   i += 256) sb2[i] = eb2[i];
  for (int i = tid; i < 1280; i += 256) sw3[i] = ew3[i];
  for (int i = tid; i < 20;   i += 256) sb3[i] = eb3[i];
  __syncthreads();

  const int row = blockIdx.x * 256 + tid;
  if (row >= BATCH) return;

  float lab[10];
#pragma unroll
  for (int i = 0; i < 10; ++i) lab[i] = labels[row * 10 + i];

  float h1[32];
#pragma unroll
  for (int j = 0; j < 32; ++j) {
    float s = sb1[j];
#pragma unroll
    for (int i = 0; i < 10; ++i) s += sw1[j * 10 + i] * lab[i];
    h1[j] = (s > 0.f) ? s : 0.2f * s;
  }
  float h2[64];
#pragma unroll
  for (int j = 0; j < 64; ++j) {
    float s = sb2[j];
#pragma unroll
    for (int i = 0; i < 32; ++i) s += sw2[j * 32 + i] * h1[i];
    h2[j] = (s > 0.f) ? s : 0.2f * s;
  }
  float* arow = angles + row * 20;
#pragma unroll
  for (int j = 0; j < 20; ++j) {
    float s = sb3[j];
#pragma unroll
    for (int i = 0; i < 64; ++i) s += sw3[j * 64 + i] * h2[i];
    arow[j] = tanhf(s) * PI_F;
  }

  _Float16* xr = xpad + (size_t)row * KPAD;
#pragma unroll
  for (int i = 0; i < 10; ++i) xr[1024 + i] = (_Float16)lab[i];
#pragma unroll
  for (int i = 0; i < 22; ++i) xr[1034 + i] = (_Float16)0.f;
}

// ---------------------------------------------------------------------------
// Weight pack: f32 -> f16, pad W1 K-dim 1034 -> 1056 with zeros.
// ---------------------------------------------------------------------------
__global__ void packw_kernel(const float* __restrict__ pw1,
                             const float* __restrict__ pw2,
                             const float* __restrict__ pw3,
                             _Float16* __restrict__ W1,
                             _Float16* __restrict__ W2,
                             _Float16* __restrict__ W3)
{
  const int i = blockIdx.x * blockDim.x + threadIdx.x;
  const int n1 = 128 * KPAD, n2 = 128 * 128, n3 = 64 * 128;
  if (i < n1) {
    int r = i / KPAD, c = i % KPAD;
    W1[i] = (_Float16)((c < 1034) ? pw1[r * 1034 + c] : 0.f);
  } else if (i < n1 + n2) {
    int j = i - n1;
    W2[j] = (_Float16)pw2[j];
  } else if (i < n1 + n2 + n3) {
    int j = i - n1 - n2;
    W3[j] = (_Float16)pw3[j];
  }
}

// ---------------------------------------------------------------------------
// GEMM: out(MxN) = act(A(MxK,f16) @ W(NxK,f16)^T + bias), f32 accumulate via
// v_wmma_f32_16x16x32_f16. One wave per 16x16 output tile. K % 32 == 0.
// A per-lane layout: M=lane%16,, halves 0..7 -> K (0..7 | 8..15 by lane-hi),
// halves 8..15 -> K+16. B per-lane: N=lane%16, same K layout as A.
// act: 0 = LeakyReLU(0.2), 1 = tanh.
// ---------------------------------------------------------------------------
__global__ __launch_bounds__(256) void gemm_kernel(
    const _Float16* __restrict__ A, int lda,
    const _Float16* __restrict__ W, int ldw,
    const float* __restrict__ bias,
    float* __restrict__ out, int M, int N, int K, int act)
{
  const int wave = threadIdx.x >> 5;
  const int lane = threadIdx.x & 31;
  const int tile = blockIdx.x * 8 + wave;
  const int tilesN = N >> 4;
  if (tile >= (M >> 4) * tilesN) return;
  const int tm = tile / tilesN;
  const int tn = tile % tilesN;
  const int l16 = lane & 15, hi = lane >> 4;

  const _Float16* arow = A + (size_t)(tm * 16 + l16) * lda + hi * 8;
  const _Float16* brow = W + (size_t)(tn * 16 + l16) * ldw + hi * 8;

  v8f acc = {};
  for (int kb = 0; kb < K; kb += 32) {
    if (kb + 32 < K) __builtin_prefetch(arow + kb + 128, 0, 3);
    v8h a0 = *(const v8h*)(arow + kb);
    v8h a1 = *(const v8h*)(arow + kb + 16);
    v8h b0 = *(const v8h*)(brow + kb);
    v8h b1 = *(const v8h*)(brow + kb + 16);
    v16h av = __builtin_shufflevector(a0, a1, 0, 1, 2, 3, 4, 5, 6, 7,
                                      8, 9, 10, 11, 12, 13, 14, 15);
    v16h bv = __builtin_shufflevector(b0, b1, 0, 1, 2, 3, 4, 5, 6, 7,
                                      8, 9, 10, 11, 12, 13, 14, 15);
    acc = __builtin_amdgcn_wmma_f32_16x16x32_f16(
        false, av, false, bv, (short)0, acc, false, false);
  }

  const float bv = bias[tn * 16 + l16];
  float* obase = out + (size_t)(tm * 16 + hi * 8) * N + tn * 16 + l16;
#pragma unroll
  for (int r = 0; r < 8; ++r) {
    float v = acc[r] + bv;
    v = act ? tanhf(v) : ((v > 0.f) ? v : 0.2f * v);
    obase[(size_t)r * N] = v;
  }
}

// ---------------------------------------------------------------------------
// BatchNorm (training mode, biased var): stats then apply (-> f16 for next GEMM)
// ---------------------------------------------------------------------------
__global__ __launch_bounds__(256) void bnstats_kernel(
    const float* __restrict__ Y, int M, int N,
    float* __restrict__ mu, float* __restrict__ inv)
{
  __shared__ float ssum[256], ssq[256];
  const int c = blockIdx.x;
  const int tid = threadIdx.x;
  float s = 0.f, q = 0.f;
  for (int r = tid; r < M; r += 256) {
    float v = Y[(size_t)r * N + c];
    s += v;
    q += v * v;
  }
  ssum[tid] = s; ssq[tid] = q;
  __syncthreads();
  for (int o = 128; o > 0; o >>= 1) {
    if (tid < o) { ssum[tid] += ssum[tid + o]; ssq[tid] += ssq[tid + o]; }
    __syncthreads();
  }
  if (tid == 0) {
    float mean = ssum[0] / (float)M;
    float var  = ssq[0] / (float)M - mean * mean;
    mu[c]  = mean;
    inv[c] = 1.0f / sqrtf(var + 1e-5f);
  }
}

__global__ __launch_bounds__(256) void bnapply_kernel(
    const float* __restrict__ Y, const float* __restrict__ mu,
    const float* __restrict__ inv, const float* __restrict__ gma,
    const float* __restrict__ bta, _Float16* __restrict__ H, int total, int N)
{
  const int i = blockIdx.x * blockDim.x + threadIdx.x;
  if (i < total) {
    const int c = i & (N - 1);
    H[i] = (_Float16)(gma[c] * (Y[i] - mu[c]) * inv[c] + bta[c]);
  }
}

// ---------------------------------------------------------------------------
extern "C" void kernel_launch(void* const* d_in, const int* in_sizes, int n_in,
                              void* d_out, int out_size, void* d_ws,
                              size_t ws_size, hipStream_t stream)
{
  const float* noise   = (const float*)d_in[0];
  const float* labels  = (const float*)d_in[1];
  const float* qparams = (const float*)d_in[2];
  const float* ew1 = (const float*)d_in[3];  const float* eb1 = (const float*)d_in[4];
  const float* ew2 = (const float*)d_in[5];  const float* eb2 = (const float*)d_in[6];
  const float* ew3 = (const float*)d_in[7];  const float* eb3 = (const float*)d_in[8];
  const float* pw1 = (const float*)d_in[9];  const float* pb1 = (const float*)d_in[10];
  const float* g1  = (const float*)d_in[11]; const float* be1 = (const float*)d_in[12];
  const float* pw2 = (const float*)d_in[13]; const float* pb2 = (const float*)d_in[14];
  const float* g2  = (const float*)d_in[15]; const float* be2 = (const float*)d_in[16];
  const float* pw3 = (const float*)d_in[17]; const float* pb3 = (const float*)d_in[18];

  char* ws = (char*)d_ws;
  float*    angles = (float*)(ws + 0);              // 2048*20*4      = 163840
  _Float16* Xpad   = (_Float16*)(ws + 163840);      // 2048*1056*2    = 4325376
  _Float16* W1h    = (_Float16*)(ws + 4489216);     // 128*1056*2     = 270336
  _Float16* W2h    = (_Float16*)(ws + 4759552);     // 128*128*2      = 32768
  _Float16* W3h    = (_Float16*)(ws + 4792320);     // 64*128*2       = 16384
  float*    Y      = (float*)(ws + 4808704);        // 2048*128*4     = 1048576
  _Float16* H      = (_Float16*)(ws + 5857280);     // 2048*128*2     = 524288
  float*    mu1    = (float*)(ws + 6381568);
  float*    inv1   = (float*)(ws + 6382080);
  float*    mu2    = (float*)(ws + 6382592);
  float*    inv2   = (float*)(ws + 6383104);        // end ~6.09 MB

  packw_kernel<<<624, 256, 0, stream>>>(pw1, pw2, pw3, W1h, W2h, W3h);
  encoder_kernel<<<8, 256, 0, stream>>>(labels, ew1, eb1, ew2, eb2, ew3, eb3,
                                        angles, Xpad);
  circuit_kernel<<<1024, 256, 0, stream>>>(noise, angles, qparams, Xpad);

  gemm_kernel<<<128, 256, 0, stream>>>(Xpad, KPAD, W1h, KPAD, pb1, Y,
                                       BATCH, 128, KPAD, 0);
  bnstats_kernel<<<128, 256, 0, stream>>>(Y, BATCH, 128, mu1, inv1);
  bnapply_kernel<<<1024, 256, 0, stream>>>(Y, mu1, inv1, g1, be1, H,
                                           BATCH * 128, 128);

  gemm_kernel<<<128, 256, 0, stream>>>(H, 128, W2h, 128, pb2, Y,
                                       BATCH, 128, 128, 0);
  bnstats_kernel<<<128, 256, 0, stream>>>(Y, BATCH, 128, mu2, inv2);
  bnapply_kernel<<<1024, 256, 0, stream>>>(Y, mu2, inv2, g2, be2, H,
                                           BATCH * 128, 128);

  gemm_kernel<<<64, 256, 0, stream>>>(H, 128, W3h, 128, pb3, (float*)d_out,
                                      BATCH, 64, 128, 1);
}
